// MultiHeadAttention_12463995093495
// MI455X (gfx1250) — matrix operand order repaired and checked by reading, not compiled, not run on hardware
//
#include <hip/hip_runtime.h>
#include <hip/hip_bf16.h>

// ---------------------------------------------------------------------------
// MI455X (gfx1250) multi-head attention, bf16 WMMA + async-to-LDS pipelines.
// ---------------------------------------------------------------------------

typedef __attribute__((ext_vector_type(16))) __bf16 v16bf;
typedef __attribute__((ext_vector_type(8)))  float  v8f;

#define D_MODEL   1024
#define N_HEADS   16
#define HEAD_DIM  256
#define BATCH     2
#define SEQ       2048
#define M_ROWS    (BATCH * SEQ)          // 4096
#define QKV_COLS  (N_HEADS * HEAD_DIM)   // 4096

union FragB16 { v16bf v; unsigned int u[8]; uint4 q[2]; };

__device__ __forceinline__ unsigned short f32_to_bf16(float f) {
  union { float f; unsigned int u; } c; c.f = f;
  unsigned int u = c.u;
  u += 0x7FFFu + ((u >> 16) & 1u);   // round-to-nearest-even
  return (unsigned short)(u >> 16);
}

__device__ __forceinline__ v8f v8f_zero() {
  v8f z;
#pragma unroll
  for (int i = 0; i < 8; ++i) z[i] = 0.0f;
  return z;
}

// CDNA5 async DMA: global -> LDS, 16 bytes per lane, tracked by ASYNCcnt.
// lds_off is the wave-relative LDS byte offset (= low 32 bits of the generic
// pointer per the ISA aperture rule).
__device__ __forceinline__ void async_copy_b128(unsigned int lds_off,
                                                const void* gptr) {
  asm volatile("global_load_async_to_lds_b128 %0, %1, off"
               :: "v"(lds_off), "v"((unsigned long long)(uintptr_t)gptr)
               : "memory");
}

__device__ __forceinline__ void wait_async0() {
  asm volatile("s_wait_asynccnt 0x0" ::: "memory");
}

// ---------------------------------------------------------------------------
// Stage 0: fp32 -> bf16 conversion (optionally transposing weights to N-major)
// ---------------------------------------------------------------------------
__global__ void cvt_bf16_kernel(const float* __restrict__ in,
                                unsigned short* __restrict__ out, int n) {
  int i = blockIdx.x * blockDim.x + threadIdx.x;
  if (i < n) out[i] = f32_to_bf16(in[i]);
}

// in: (K rows x N cols) f32 row-major; out: (N x K) bf16 row-major
__global__ void tcvt_bf16_kernel(const float* __restrict__ in,
                                 unsigned short* __restrict__ out, int K, int N) {
  int i = blockIdx.x * blockDim.x + threadIdx.x;
  if (i >= K * N) return;
  int k = i / N, n = i % N;
  out[n * K + k] = f32_to_bf16(in[i]);
}

// ---------------------------------------------------------------------------
// Tiled GEMM with async-to-LDS double buffering.
//   C(MxN) = A(MxK) @ B(KxN), B given N-major as BT(NxK), all bf16.
// Workgroup (128 thr = 4 waves) -> 128x64 tile; wave w -> rows w*32..w*32+31.
// Per k-step/wave: 12 ds_load_b128 -> 8 WMMAs (fragments hoisted, single wait).
// mode 0: write bf16 (B,H,S,d)   (Q, K projections)
// mode 1: write bf16 (B,H,d,S)   (V transposed)
// mode 2: write f32 row-major    (final output)
// ---------------------------------------------------------------------------
__global__ __launch_bounds__(128) void tile_gemm_kernel(
    const unsigned short* __restrict__ A,
    const unsigned short* __restrict__ BT,
    void* __restrict__ Out,
    int M, int N, int K, int mode) {
  __shared__ unsigned short Abuf[2][128 * 32];  // 2 x 8KB
  __shared__ unsigned short Bbuf[2][64 * 32];   // 2 x 4KB

  const int tid  = threadIdx.x;
  const int lane = tid & 31;
  const int warp = tid >> 5;
  const int nb64 = N >> 6;
  const int m0 = (blockIdx.x / nb64) * 128;
  const int n0 = (blockIdx.x % nb64) * 64;
  const int half = lane >> 4, l16 = lane & 15;

  v8f acc[8];
#pragma unroll
  for (int t = 0; t < 8; ++t) acc[t] = v8f_zero();

  // stage one 128x32 A tile + 64x32 B tile into buffer `buf`
  auto stage = [&](int buf, int k0) {
    const unsigned int abase = (unsigned int)(uintptr_t)(&Abuf[buf][0]);
    const unsigned int bbase = (unsigned int)(uintptr_t)(&Bbuf[buf][0]);
#pragma unroll
    for (int i = 0; i < 4; ++i) {
      const int c   = tid + i * 128;      // chunk 0..511, 16B each
      const int row = c >> 2;             // tile row 0..127
      const int ke  = (c & 3) * 8;        // element offset in the 32-wide k slab
      async_copy_b128(abase + c * 16, A + (size_t)(m0 + row) * K + k0 + ke);
    }
#pragma unroll
    for (int i = 0; i < 2; ++i) {
      const int c   = tid + i * 128;      // chunk 0..255
      const int row = c >> 2;             // tile row 0..63
      const int ke  = (c & 3) * 8;
      async_copy_b128(bbase + c * 16, BT + (size_t)(n0 + row) * K + k0 + ke);
    }
  };

  stage(0, 0);
  wait_async0();
  __syncthreads();

  int buf = 0;
  for (int k0 = 0; k0 < K; k0 += 32) {
    if (k0 + 32 < K) stage(buf ^ 1, k0 + 32);   // overlap copy with compute

    // hoist ALL fragments first -> one dscnt wait, then 8 back-to-back WMMAs
    FragB16 af[2];
#pragma unroll
    for (int s = 0; s < 2; ++s) {
      const unsigned short* ar =
          &Abuf[buf][(warp * 32 + s * 16 + l16) * 32 + half * 8];
      af[s].q[0] = *(const uint4*)(ar);
      af[s].q[1] = *(const uint4*)(ar + 16);
    }
    FragB16 bfr[4];
#pragma unroll
    for (int t = 0; t < 4; ++t) {
      const unsigned short* br = &Bbuf[buf][(t * 16 + l16) * 32 + half * 8];
      bfr[t].q[0] = *(const uint4*)(br);
      bfr[t].q[1] = *(const uint4*)(br + 16);
    }
#pragma unroll
    for (int s = 0; s < 2; ++s)
#pragma unroll
      for (int t = 0; t < 4; ++t)
        acc[s * 4 + t] = __builtin_amdgcn_wmma_f32_16x16x32_bf16(
            false, af[s].v, false, bfr[t].v, (short)0, acc[s * 4 + t], false, false);

    wait_async0();        // next buffer landed (our own ASYNCcnt)
    __syncthreads();      // everyone done reading `buf`, everyone's copies done
    buf ^= 1;
  }

  // ---------------- epilogue ----------------
  if (mode == 2) {
    float* O = (float*)Out;
#pragma unroll
    for (int s = 0; s < 2; ++s) {
      const int wm0 = m0 + warp * 32 + s * 16;
#pragma unroll
      for (int t = 0; t < 4; ++t) {
        const int col = n0 + t * 16 + l16;
#pragma unroll
        for (int r = 0; r < 8; ++r)
          O[(size_t)(wm0 + r + 8 * half) * N + col] = acc[s * 4 + t][r];
      }
    }
  } else {
    unsigned short* O = (unsigned short*)Out;
#pragma unroll
    for (int s = 0; s < 2; ++s) {
      const int wm0 = m0 + warp * 32 + s * 16;
#pragma unroll
      for (int t = 0; t < 4; ++t) {
        const int col = n0 + t * 16 + l16;
        const int h = col >> 8, dd = col & 255;
#pragma unroll
        for (int r = 0; r < 8; ++r) {
          const int row = wm0 + r + 8 * half;
          const int b = row >> 11, s2 = row & 2047;
          size_t idx;
          if (mode == 1)
            idx = (((size_t)(b * N_HEADS + h)) * HEAD_DIM + dd) * SEQ + s2;
          else
            idx = (((size_t)(b * N_HEADS + h)) * SEQ + s2) * HEAD_DIM + dd;
          O[idx] = f32_to_bf16(acc[s * 4 + t][r]);
        }
      }
    }
  }
}

// ---------------------------------------------------------------------------
// Causal flash attention. One wave per 16-query tile. Bk = 32.
// Q,K: (B,H,S,d) bf16.  VT: (B,H,d,S) bf16.  Y: (B*S, H*d) bf16.
// ---------------------------------------------------------------------------
__global__ __launch_bounds__(128) void flash_attn_kernel(
    const unsigned short* __restrict__ Q,
    const unsigned short* __restrict__ Kb,
    const unsigned short* __restrict__ VT,
    unsigned short* __restrict__ Y) {
  const int lane = threadIdx.x & 31;
  const int warp = threadIdx.x >> 5;
  const int wave = blockIdx.x * (blockDim.x >> 5) + warp;
  const int qTiles = SEQ / 16;                      // 128
  const int qt = wave % qTiles;
  const int bh = wave / qTiles;                     // 0..31
  const int h = bh % N_HEADS, b = bh / N_HEADS;
  const int q0 = qt * 16;
  const int half = lane >> 4, l16 = lane & 15;

  const unsigned short* Qh = Q  + (size_t)bh * SEQ * HEAD_DIM;
  const unsigned short* Kh = Kb + (size_t)bh * SEQ * HEAD_DIM;
  const unsigned short* Vh = VT + (size_t)bh * HEAD_DIM * SEQ;

  __shared__ unsigned short plds[4][16 * 32];
  unsigned short* pl = plds[warp];

  // Q tile 16x256 as 8 A-fragments (K-chunks of 32), resident in VGPRs
  FragB16 qf[8];
  const unsigned short* qrow = Qh + (size_t)(q0 + l16) * HEAD_DIM;
#pragma unroll
  for (int c = 0; c < 8; ++c) {
    const unsigned short* qr = qrow + c * 32 + half * 8;
    qf[c].q[0] = *(const uint4*)(qr);
    qf[c].q[1] = *(const uint4*)(qr + 16);
  }

  v8f acc[16];
#pragma unroll
  for (int n = 0; n < 16; ++n) acc[n] = v8f_zero();

  float m[8], l[8];
#pragma unroll
  for (int r = 0; r < 8; ++r) { m[r] = -1e30f; l[r] = 0.0f; }

  const float scale = 0.0625f;                      // 1/sqrt(256)
  const int ktMax = (q0 + 15) / 32;                 // inclusive, causal

  for (int kt = 0; kt <= ktMax; ++kt) {
    const int j0 = kt * 32;

    // S = Q @ K^T : two 16x16 f32 tiles (keys j0..j0+15, j0+16..j0+31)
    v8f sc0 = v8f_zero(), sc1 = v8f_zero();
#pragma unroll
    for (int c = 0; c < 8; ++c) {
      FragB16 kf0, kf1;
      const int cb = c * 32 + half * 8;
      const unsigned short* kr0 = Kh + (size_t)(j0 + l16)      * HEAD_DIM + cb;
      const unsigned short* kr1 = Kh + (size_t)(j0 + 16 + l16) * HEAD_DIM + cb;
      kf0.q[0] = *(const uint4*)(kr0);
      kf0.q[1] = *(const uint4*)(kr0 + 16);
      kf1.q[0] = *(const uint4*)(kr1);
      kf1.q[1] = *(const uint4*)(kr1 + 16);
      sc0 = __builtin_amdgcn_wmma_f32_16x16x32_bf16(
          false, qf[c].v, false, kf0.v, (short)0, sc0, false, false);
      sc1 = __builtin_amdgcn_wmma_f32_16x16x32_bf16(
          false, qf[c].v, false, kf1.v, (short)0, sc1, false, false);
    }

    // online softmax over the 32 new columns
#pragma unroll
    for (int r = 0; r < 8; ++r) {
      const int row = q0 + r + 8 * half;
      float x0 = sc0[r] * scale;
      float x1 = sc1[r] * scale;
      if (j0 + l16 > row)      x0 = -1e30f;
      if (j0 + 16 + l16 > row) x1 = -1e30f;
      float mx = fmaxf(x0, x1);
#pragma unroll
      for (int off = 1; off < 16; off <<= 1)
        mx = fmaxf(mx, __shfl_xor(mx, off, 32));    // reduce within 16-lane half
      const float mn = fmaxf(m[r], mx);
      const float p0 = __expf(x0 - mn);
      const float p1 = __expf(x1 - mn);
      float rs = p0 + p1;
#pragma unroll
      for (int off = 1; off < 16; off <<= 1)
        rs += __shfl_xor(rs, off, 32);
      const float alpha = __expf(m[r] - mn);
      m[r] = mn;
      l[r] = l[r] * alpha + rs;
#pragma unroll
      for (int n = 0; n < 16; ++n) acc[n][r] *= alpha;
      // stage P tile (16x32) in LDS for the A-fragment reload
      pl[(r + 8 * half) * 32 + l16]      = f32_to_bf16(p0);
      pl[(r + 8 * half) * 32 + 16 + l16] = f32_to_bf16(p1);
    }

    // P (16x32) as one bf16 A-fragment from LDS (wave-private slab)
    FragB16 pf;
    {
      const unsigned short* pr = pl + l16 * 32 + half * 8;
      pf.q[0] = *(const uint4*)(pr);
      pf.q[1] = *(const uint4*)(pr + 16);
    }

    // acc += P @ V  (V^T layout makes B-fragment loads contiguous along S)
#pragma unroll
    for (int n = 0; n < 16; ++n) {
      FragB16 vf;
      const unsigned short* vcol =
          Vh + (size_t)(n * 16 + l16) * SEQ + j0 + half * 8;
      vf.q[0] = *(const uint4*)(vcol);
      vf.q[1] = *(const uint4*)(vcol + 16);
      acc[n] = __builtin_amdgcn_wmma_f32_16x16x32_bf16(
          false, pf.v, false, vf.v, (short)0, acc[n], false, false);
    }
  }

  // epilogue: y = acc / l, write bf16 into (B*S, H*d)
#pragma unroll
  for (int r = 0; r < 8; ++r) {
    const float inv = 1.0f / l[r];
    const int row = q0 + r + 8 * half;
    unsigned short* yrow =
        Y + ((size_t)b * SEQ + row) * QKV_COLS + h * HEAD_DIM;
#pragma unroll
    for (int n = 0; n < 16; ++n)
      yrow[n * 16 + l16] = f32_to_bf16(acc[n][r] * inv);
  }
}

// ---------------------------------------------------------------------------
extern "C" void kernel_launch(void* const* d_in, const int* in_sizes, int n_in,
                              void* d_out, int out_size, void* d_ws, size_t ws_size,
                              hipStream_t stream) {
  (void)in_sizes; (void)n_in; (void)out_size; (void)ws_size;
  const float* x  = (const float*)d_in[0];
  // d_in[1] = causal mask (unused: mask applied analytically)
  const float* Wq = (const float*)d_in[2];
  const float* Wk = (const float*)d_in[3];
  const float* Wv = (const float*)d_in[4];
  const float* Wo = (const float*)d_in[5];
  float* out = (float*)d_out;

  char* ws = (char*)d_ws;
  size_t off = 0;
  auto alloc = [&](size_t bytes) -> char* {
    char* p = ws + off;
    off += (bytes + 255) & ~((size_t)255);
    return p;
  };
  unsigned short* xb  = (unsigned short*)alloc((size_t)M_ROWS   * D_MODEL * 2);
  unsigned short* wqT = (unsigned short*)alloc((size_t)QKV_COLS * D_MODEL * 2);
  unsigned short* wkT = (unsigned short*)alloc((size_t)QKV_COLS * D_MODEL * 2);
  unsigned short* wvT = (unsigned short*)alloc((size_t)QKV_COLS * D_MODEL * 2);
  unsigned short* woT = (unsigned short*)alloc((size_t)D_MODEL  * QKV_COLS * 2);
  unsigned short* Qb  = (unsigned short*)alloc((size_t)M_ROWS   * QKV_COLS * 2);
  unsigned short* Kb  = (unsigned short*)alloc((size_t)M_ROWS   * QKV_COLS * 2);
  unsigned short* VTb = (unsigned short*)alloc((size_t)M_ROWS   * QKV_COLS * 2);
  unsigned short* Yb  = (unsigned short*)alloc((size_t)M_ROWS   * QKV_COLS * 2);

  // Stage 0: precision conversion + weight transposes
  {
    const int n = M_ROWS * D_MODEL;
    cvt_bf16_kernel<<<(n + 255) / 256, 256, 0, stream>>>(x, xb, n);
    const int nw = D_MODEL * QKV_COLS;
    tcvt_bf16_kernel<<<(nw + 255) / 256, 256, 0, stream>>>(Wq, wqT, D_MODEL, QKV_COLS);
    tcvt_bf16_kernel<<<(nw + 255) / 256, 256, 0, stream>>>(Wk, wkT, D_MODEL, QKV_COLS);
    tcvt_bf16_kernel<<<(nw + 255) / 256, 256, 0, stream>>>(Wv, wvT, D_MODEL, QKV_COLS);
    tcvt_bf16_kernel<<<(nw + 255) / 256, 256, 0, stream>>>(Wo, woT, QKV_COLS, D_MODEL);
  }

  // Stage 1: Q/K/V projections. 128x64 block tiles -> (4096/128)*(4096/64)=2048
  tile_gemm_kernel<<<2048, 128, 0, stream>>>(xb, wqT, Qb,  M_ROWS, QKV_COLS, D_MODEL, 0);
  tile_gemm_kernel<<<2048, 128, 0, stream>>>(xb, wkT, Kb,  M_ROWS, QKV_COLS, D_MODEL, 0);
  tile_gemm_kernel<<<2048, 128, 0, stream>>>(xb, wvT, VTb, M_ROWS, QKV_COLS, D_MODEL, 1);

  // Stage 2: causal flash attention (4096 waves -> 1024 blocks)
  flash_attn_kernel<<<1024, 128, 0, stream>>>(Qb, Kb, VTb, Yb);

  // Stage 3: output projection. (4096/128)*(1024/64) = 512 blocks, fp32 out
  tile_gemm_kernel<<<512, 128, 0, stream>>>(Yb, woT, out, M_ROWS, D_MODEL, QKV_COLS, 2);
}
